// DAM_79182017069398
// MI455X (gfx1250) — matrix operand order repaired and checked by reading, not compiled, not run on hardware
//
#include <hip/hip_runtime.h>

// ---------------------------------------------------------------------------
// CDNA5 (gfx1250) bf16 WMMA implementation of the dual-attention module.
// All GEMM operands are stored K-contiguous; fragment loads are 16B vectors
// from lane-resolved pointers hoisted out of the K loops. Conv boundary is
// handled by pointing invalid lanes at a zeroed page (no per-iteration VALU).
// ---------------------------------------------------------------------------

typedef __attribute__((ext_vector_type(16))) __bf16 v16bf;
typedef __attribute__((ext_vector_type(8)))  __bf16 v8bf;
typedef __attribute__((ext_vector_type(8)))  float  v8f;

__device__ __forceinline__ v8f vzero8f() {
    v8f z;
#pragma unroll
    for (int i = 0; i < 8; ++i) z[i] = 0.0f;
    return z;
}
__device__ __forceinline__ v16bf combine16(v8bf lo, v8bf hi) {
    return __builtin_shufflevector(lo, hi, 0, 1, 2, 3, 4, 5, 6, 7,
                                   8, 9, 10, 11, 12, 13, 14, 15);
}

// A-fragment (16x32, M x K): caller passes lane pointer = base + row*stride + 8*half.
__device__ __forceinline__ v16bf load_a_lp(const __bf16* lp) {
    return combine16(*(const v8bf*)(lp), *(const v8bf*)(lp + 16));
}
// B-fragment (32x16, K x N): caller passes lane pointer = base + col*stride + 8*half.
__device__ __forceinline__ v16bf load_b_lp(const __bf16* lp) {
    return combine16(*(const v8bf*)(lp), *(const v8bf*)(lp + 16));
}

__device__ __forceinline__ v8f wmma_bf16(v16bf a, v16bf b, v8f c) {
    return __builtin_amdgcn_wmma_f32_16x16x32_bf16(false, a, false, b, (short)0, c,
                                                   false, false);
}

// ---------------------------------------------------------------------------
// Elementwise / repack kernels
// ---------------------------------------------------------------------------

__global__ void k_zero_bytes(float* __restrict__ p, int n4) {
    int i = blockIdx.x * blockDim.x + threadIdx.x;
    if (i < n4) p[i] = 0.0f;
}

__global__ void k_nchw_to_nhwc(const float* __restrict__ src, __bf16* __restrict__ dst,
                               int C, int S, int total) {
    int i = blockIdx.x * blockDim.x + threadIdx.x;
    if (i >= total) return;
    int c  = i % C;
    int bs = i / C;
    int s  = bs % S;
    int b  = bs / S;
    dst[i] = (__bf16)src[((size_t)(b * C + c)) * S + s];
}

// conv weight [Co][Ci][3][3] fp32 -> packed [9][Co][Ci] bf16
__global__ void k_pack_conv3(const float* __restrict__ src, __bf16* __restrict__ dst,
                             int Co, int Ci, int total) {
    int i = blockIdx.x * blockDim.x + threadIdx.x;
    if (i >= total) return;
    int ci = i % Ci;
    int co = (i / Ci) % Co;
    int t  = i / (Ci * Co);
    dst[i] = (__bf16)src[((size_t)(co * Ci + ci)) * 9 + t];
}

__global__ void k_cvt_bf16(const float* __restrict__ src, __bf16* __restrict__ dst, int n) {
    int i = blockIdx.x * blockDim.x + threadIdx.x;
    if (i < n) dst[i] = (__bf16)src[i];
}

// NHWC [b][s][C] bf16 -> [b][C][s] bf16
__global__ void k_transpose(const __bf16* __restrict__ src, __bf16* __restrict__ dst,
                            int C, int S, int total) {
    int i = blockIdx.x * blockDim.x + threadIdx.x;
    if (i >= total) return;
    int s = i % S;
    int c = (i / S) % C;
    int b = i / (S * C);
    dst[i] = src[((size_t)(b * S + s)) * C + c];
}

// ---------------------------------------------------------------------------
// 3x3 conv implicit GEMM: D[s, co] = sum_t sum_ci X[s(t), ci] * W[t][co][ci]
// wg: 64 spatial (4 waves x 16) x 128 co (8 wmma tiles per wave).
// W-boundary: invalid lanes read a zeroed page (address select, hoisted).
// mode 0: BN+ReLU -> bf16 NHWC ; mode 1: raw fp32 NCHW (d_out)
// ---------------------------------------------------------------------------
__global__ void k_conv3(const __bf16* __restrict__ xin0, int cin0,
                        const __bf16* __restrict__ xin1, int cin1,
                        const __bf16* __restrict__ wp, int Cout, int mode,
                        const __bf16* __restrict__ zpage,
                        const float* __restrict__ bn_sc, const float* __restrict__ bn_bi,
                        const float* __restrict__ bn_mn, const float* __restrict__ bn_vr,
                        __bf16* __restrict__ out_bf, float* __restrict__ out_f32) {
    const int lane = threadIdx.x & 31;
    const int wave = threadIdx.x >> 5;
    const int row  = lane & 15;
    const int half = lane >> 4;
    const int co0  = blockIdx.x * 128;
    const int nh   = blockIdx.y;  // n*64 + h
    const int n    = nh >> 6, h = nh & 63;
    const int w0   = wave * 16;
    const int Cin  = cin0 + cin1;

    const __bf16* wlane = wp + ((size_t)(co0 + row)) * Cin + 8 * half;
    const size_t  jstep = (size_t)16 * Cin;   // B sub-tile stride
    const size_t  tstep = (size_t)Cout * Cin; // tap stride

    v8f acc[8];
#pragma unroll
    for (int j = 0; j < 8; ++j) acc[j] = vzero8f();

    for (int t = 0; t < 9; ++t) {
        const int dh = t / 3 - 1, dw = t % 3 - 1;
        const int hh = h + dh;
        if ((unsigned)hh >= 64u) continue;
        const int  wl    = w0 + dw + row;
        const bool valid = (unsigned)wl < 64u;
        const int  sp    = (n * 64 + hh) * 64 + (valid ? wl : 0);
        const __bf16* wt = wlane + (size_t)t * tstep;

        int kglob = 0;
#pragma unroll
        for (int part = 0; part < 2; ++part) {
            const __bf16* xp = part ? xin1 : xin0;
            const int     Cp = part ? cin1 : cin0;
            if (Cp == 0) continue;
            const __bf16* apReal = xp + (size_t)sp * Cp + 8 * half;
            const __bf16* ap     = valid ? apReal : zpage;  // one 64-bit select/tap
            const __bf16* bp     = wt + kglob;
            for (int kk = 0; kk < Cp; kk += 64) {
                // half 0
                v16bf a0 = load_a_lp(ap + kk);
                v16bf b0[8];
#pragma unroll
                for (int j = 0; j < 8; ++j)
                    b0[j] = load_b_lp(bp + (size_t)j * jstep + kk);
#pragma unroll
                for (int j = 0; j < 8; ++j) acc[j] = wmma_bf16(a0, b0[j], acc[j]);
                // half 1 (distinct A registers; avoids WMMA->VALU reuse hazard)
                v16bf a1 = load_a_lp(ap + kk + 32);
                v16bf b1[8];
#pragma unroll
                for (int j = 0; j < 8; ++j)
                    b1[j] = load_b_lp(bp + (size_t)j * jstep + kk + 32);
#pragma unroll
                for (int j = 0; j < 8; ++j) acc[j] = wmma_bf16(a1, b1[j], acc[j]);
            }
            kglob += Cp;
        }
    }

    if (mode == 0) {
#pragma unroll
        for (int j = 0; j < 8; ++j) {
            const int co     = co0 + 16 * j + row;
            const float inv  = bn_sc[co] * rsqrtf(bn_vr[co] + 1e-5f);
            const float beta = bn_bi[co] - bn_mn[co] * inv;
#pragma unroll
            for (int r = 0; r < 8; ++r) {
                const int wlo = w0 + r + 8 * half;
                float v = acc[j][r] * inv + beta;
                v = v > 0.0f ? v : 0.0f;
                out_bf[((size_t)((n * 64 + h) * 64 + wlo)) * Cout + co] = (__bf16)v;
            }
        }
    } else {
#pragma unroll
        for (int j = 0; j < 8; ++j) {
            const int co = co0 + 16 * j + row;
#pragma unroll
            for (int r = 0; r < 8; ++r) {
                const int wlo = w0 + r + 8 * half;
                out_f32[((size_t)(n * Cout + co)) * 4096 + h * 64 + wlo] = acc[j][r];
            }
        }
    }
}

// ---------------------------------------------------------------------------
// 1x1 conv GEMM: D[s, co] = X[s,:].W[co,:]; wg: 64 s x 128 co.
// mode 0: bf16 [s][Cout] ; mode 1: transposed bf16 [b][Cout][4096]
// ---------------------------------------------------------------------------
__global__ void k_gemm1(const __bf16* __restrict__ X, const __bf16* __restrict__ W,
                        int Cout, int mode, __bf16* __restrict__ out) {
    const int lane = threadIdx.x & 31;
    const int wave = threadIdx.x >> 5;
    const int row  = lane & 15;
    const int half = lane >> 4;
    const int m0   = blockIdx.y * 64 + wave * 16;
    const int co0  = blockIdx.x * 128;

    const __bf16* ap = X + (size_t)(m0 + row) * 512 + 8 * half;
    const __bf16* bp = W + (size_t)(co0 + row) * 512 + 8 * half;

    v8f acc[8];
#pragma unroll
    for (int j = 0; j < 8; ++j) acc[j] = vzero8f();

    for (int kk = 0; kk < 512; kk += 64) {
        v16bf a0 = load_a_lp(ap + kk);
        v16bf b0[8];
#pragma unroll
        for (int j = 0; j < 8; ++j) b0[j] = load_b_lp(bp + (size_t)j * 16 * 512 + kk);
#pragma unroll
        for (int j = 0; j < 8; ++j) acc[j] = wmma_bf16(a0, b0[j], acc[j]);
        v16bf a1 = load_a_lp(ap + kk + 32);
        v16bf b1[8];
#pragma unroll
        for (int j = 0; j < 8; ++j) b1[j] = load_b_lp(bp + (size_t)j * 16 * 512 + kk + 32);
#pragma unroll
        for (int j = 0; j < 8; ++j) acc[j] = wmma_bf16(a1, b1[j], acc[j]);
    }

#pragma unroll
    for (int j = 0; j < 8; ++j) {
        const int co = co0 + 16 * j + row;
#pragma unroll
        for (int r = 0; r < 8; ++r) {
            const int s = m0 + r + 8 * half;
            if (mode == 0) {
                out[(size_t)s * Cout + co] = (__bf16)acc[j][r];
            } else {
                const int b = s >> 12, sl = s & 4095;
                out[((size_t)(b * Cout + co)) * 4096 + sl] = (__bf16)acc[j][r];
            }
        }
    }
}

// ---------------------------------------------------------------------------
// SAM pass 1: online softmax row stats of S = (q.kT)/8
// qk: [b*4096][128] bf16; q = ch 0..63, k = ch 64..127
// ---------------------------------------------------------------------------
__global__ void k_sam_p1(const __bf16* __restrict__ qk, float* __restrict__ rmax,
                         float* __restrict__ rsum) {
    const int lane = threadIdx.x & 31;
    const int wave = threadIdx.x >> 5;
    const int row  = lane & 15;
    const int half = lane >> 4;
    const int b    = blockIdx.z;
    const int s0   = blockIdx.x * 64 + wave * 16;

    const __bf16* qlp = qk + ((size_t)(b * 4096 + s0 + row)) * 128 + 8 * half;
    const __bf16* klp = qk + ((size_t)(b * 4096 + row)) * 128 + 64 + 8 * half;

    v16bf aq0 = load_a_lp(qlp);
    v16bf aq1 = load_a_lp(qlp + 32);

    float mi[8], li[8];
#pragma unroll
    for (int r = 0; r < 8; ++r) { mi[r] = -3.0e38f; li[r] = 0.0f; }

    for (int tt = 0; tt < 4096; tt += 16) {
        const __bf16* kp = klp + (size_t)tt * 128;
        v16bf b0 = load_b_lp(kp);
        v16bf b1 = load_b_lp(kp + 32);
        v8f s = vzero8f();
        s = wmma_bf16(aq0, b0, s);
        s = wmma_bf16(aq1, b1, s);
#pragma unroll
        for (int r = 0; r < 8; ++r) {
            const float sv = s[r] * 0.125f;
            float rm = sv;
#pragma unroll
            for (int off = 1; off < 16; off <<= 1)
                rm = fmaxf(rm, __shfl_xor(rm, off, 16));
            const float mnew = fmaxf(mi[r], rm);
            float pe = __expf(sv - mnew);
#pragma unroll
            for (int off = 1; off < 16; off <<= 1)
                pe += __shfl_xor(pe, off, 16);
            li[r] = li[r] * __expf(mi[r] - mnew) + pe;
            mi[r] = mnew;
        }
    }
    if (row == 0) {
#pragma unroll
        for (int r = 0; r < 8; ++r) {
            const int srow = b * 4096 + s0 + r + 8 * half;
            rmax[srow] = mi[r];
            rsum[srow] = li[r];
        }
    }
}

// ---------------------------------------------------------------------------
// SAM pass 2: O[s,c] = sum_t P[s,t]*vT[c][t]; out = gamma*O + xs_bn
// wg: 16 s rows; each wave owns a 128-channel strip; P staged through LDS.
// ---------------------------------------------------------------------------
__global__ void k_sam_p2(const __bf16* __restrict__ qk, const __bf16* __restrict__ vT,
                         const float* __restrict__ rmax, const float* __restrict__ rsum,
                         const __bf16* __restrict__ xs_bn, const float* __restrict__ gamma,
                         __bf16* __restrict__ xs_out) {
    __shared__ __bf16 lP[4][16][32];
    const int lane = threadIdx.x & 31;
    const int wave = threadIdx.x >> 5;
    const int row  = lane & 15;
    const int half = lane >> 4;
    const int b    = blockIdx.z;
    const int s0   = blockIdx.x * 16;
    const int c0   = wave * 128;

    const __bf16* qlp = qk + ((size_t)(b * 4096 + s0 + row)) * 128 + 8 * half;
    const __bf16* klp = qk + ((size_t)(b * 4096 + row)) * 128 + 64 + 8 * half;
    const __bf16* vlp = vT + ((size_t)(b * 512 + c0 + row)) * 4096 + 8 * half;
    const __bf16* plp = &lP[wave][row][0] + 8 * half;

    v16bf aq0 = load_a_lp(qlp);
    v16bf aq1 = load_a_lp(qlp + 32);

    float m8[8], il8[8];
#pragma unroll
    for (int r = 0; r < 8; ++r) {
        const int srow = b * 4096 + s0 + r + 8 * half;
        m8[r]  = rmax[srow];
        il8[r] = 1.0f / rsum[srow];
    }

    v8f acc[8];
#pragma unroll
    for (int j = 0; j < 8; ++j) acc[j] = vzero8f();

    for (int tt = 0; tt < 4096; tt += 32) {
#pragma unroll
        for (int tn = 0; tn < 32; tn += 16) {
            const __bf16* kp = klp + (size_t)(tt + tn) * 128;
            v16bf b0 = load_b_lp(kp);
            v16bf b1 = load_b_lp(kp + 32);
            v8f s = vzero8f();
            s = wmma_bf16(aq0, b0, s);
            s = wmma_bf16(aq1, b1, s);
#pragma unroll
            for (int r = 0; r < 8; ++r) {
                const float p = __expf(s[r] * 0.125f - m8[r]) * il8[r];
                lP[wave][r + 8 * half][tn + row] = (__bf16)p;
            }
        }
        v16bf ap = load_a_lp(plp);
        v16bf bfr[8];
#pragma unroll
        for (int j = 0; j < 8; ++j) bfr[j] = load_b_lp(vlp + (size_t)j * 16 * 4096 + tt);
#pragma unroll
        for (int j = 0; j < 8; ++j) acc[j] = wmma_bf16(ap, bfr[j], acc[j]);
    }

    const float g = gamma[0];
#pragma unroll
    for (int j = 0; j < 8; ++j) {
        const int co = c0 + 16 * j + row;
#pragma unroll
        for (int r = 0; r < 8; ++r) {
            const int sg = b * 4096 + s0 + r + 8 * half;
            const float o = g * acc[j][r] + (float)xs_bn[(size_t)sg * 512 + co];
            xs_out[(size_t)sg * 512 + co] = (__bf16)o;
        }
    }
}

// ---------------------------------------------------------------------------
// CAM gram: m[c,d] = (1/64) * sum_s y[c,s]*y[d,s]; wg: 16 c x 128 d per wave
// ---------------------------------------------------------------------------
__global__ void k_gram(const __bf16* __restrict__ xcT, float* __restrict__ mgram) {
    const int lane = threadIdx.x & 31;
    const int wave = threadIdx.x >> 5;
    const int row  = lane & 15;
    const int half = lane >> 4;
    const int b    = blockIdx.z;
    const int c0   = blockIdx.y * 64 + wave * 16;
    const int d0   = blockIdx.x * 128;

    const __bf16* ap = xcT + ((size_t)(b * 512 + c0 + row)) * 4096 + 8 * half;
    const __bf16* bp = xcT + ((size_t)(b * 512 + d0 + row)) * 4096 + 8 * half;

    v8f acc[8];
#pragma unroll
    for (int j = 0; j < 8; ++j) acc[j] = vzero8f();

    for (int kk = 0; kk < 4096; kk += 64) {
        v16bf a0 = load_a_lp(ap + kk);
        v16bf b0[8];
#pragma unroll
        for (int j = 0; j < 8; ++j) b0[j] = load_b_lp(bp + (size_t)j * 16 * 4096 + kk);
#pragma unroll
        for (int j = 0; j < 8; ++j) acc[j] = wmma_bf16(a0, b0[j], acc[j]);
        v16bf a1 = load_a_lp(ap + kk + 32);
        v16bf b1[8];
#pragma unroll
        for (int j = 0; j < 8; ++j) b1[j] = load_b_lp(bp + (size_t)j * 16 * 4096 + kk + 32);
#pragma unroll
        for (int j = 0; j < 8; ++j) acc[j] = wmma_bf16(a1, b1[j], acc[j]);
    }
#pragma unroll
    for (int j = 0; j < 8; ++j)
#pragma unroll
        for (int r = 0; r < 8; ++r)
            mgram[((size_t)(b * 512 + c0 + r + 8 * half)) * 512 + d0 + 16 * j + row] =
                acc[j][r] * (1.0f / 64.0f);
}

// row softmax over 512 cols: fp32 in -> bf16 P out
__global__ void k_softmax512(const float* __restrict__ m, __bf16* __restrict__ P) {
    const int row = blockIdx.x;
    const float* rp = m + (size_t)row * 512;
    __shared__ float red[256];
    const int t = threadIdx.x;
    const float a = rp[t], b = rp[t + 256];
    red[t] = fmaxf(a, b);
    __syncthreads();
    for (int s = 128; s > 0; s >>= 1) {
        if (t < s) red[t] = fmaxf(red[t], red[t + s]);
        __syncthreads();
    }
    const float mx = red[0];
    __syncthreads();
    const float e0 = __expf(a - mx), e1 = __expf(b - mx);
    red[t] = e0 + e1;
    __syncthreads();
    for (int s = 128; s > 0; s >>= 1) {
        if (t < s) red[t] += red[t + s];
        __syncthreads();
    }
    const float inv = 1.0f / red[0];
    P[(size_t)row * 512 + t]       = (__bf16)(e0 * inv);
    P[(size_t)row * 512 + t + 256] = (__bf16)(e1 * inv);
}

// CAM apply: a[c,s] = sum_d P[c,d]*y[d,s]; out[s,c] = gamma*a + xc_bn[s,c]
__global__ void k_cam_apply(const __bf16* __restrict__ P, const __bf16* __restrict__ xc_bn,
                            const float* __restrict__ gamma, __bf16* __restrict__ xc_out) {
    const int lane = threadIdx.x & 31;
    const int wave = threadIdx.x >> 5;
    const int row  = lane & 15;
    const int half = lane >> 4;
    const int b    = blockIdx.z;
    const int c0   = blockIdx.y * 64 + wave * 16;
    const int sB   = blockIdx.x * 128;

    const __bf16* ap = P + ((size_t)(b * 512 + c0 + row)) * 512 + 8 * half;
    const __bf16* bp = xc_bn + ((size_t)(b * 4096 + sB + row)) * 512 + 8 * half;

    v8f acc[8];
#pragma unroll
    for (int j = 0; j < 8; ++j) acc[j] = vzero8f();

    for (int kk = 0; kk < 512; kk += 64) {
        v16bf a0 = load_a_lp(ap + kk);
        v16bf b0[8];
#pragma unroll
        for (int j = 0; j < 8; ++j) b0[j] = load_b_lp(bp + (size_t)j * 16 * 512 + kk);
#pragma unroll
        for (int j = 0; j < 8; ++j) acc[j] = wmma_bf16(a0, b0[j], acc[j]);
        v16bf a1 = load_a_lp(ap + kk + 32);
        v16bf b1[8];
#pragma unroll
        for (int j = 0; j < 8; ++j) b1[j] = load_b_lp(bp + (size_t)j * 16 * 512 + kk + 32);
#pragma unroll
        for (int j = 0; j < 8; ++j) acc[j] = wmma_bf16(a1, b1[j], acc[j]);
    }
    const float g = gamma[0];
#pragma unroll
    for (int j = 0; j < 8; ++j)
#pragma unroll
        for (int r = 0; r < 8; ++r) {
            const int sg = b * 4096 + sB + 16 * j + row;
            const int c  = c0 + r + 8 * half;
            const float o = g * acc[j][r] + (float)xc_bn[(size_t)sg * 512 + c];
            xc_out[(size_t)sg * 512 + c] = (__bf16)o;
        }
}

// ---------------------------------------------------------------------------
// Host-side orchestration
// ---------------------------------------------------------------------------
extern "C" void kernel_launch(void* const* d_in, const int* in_sizes, int n_in,
                              void* d_out, int out_size, void* d_ws, size_t ws_size,
                              hipStream_t stream) {
    const float* x         = (const float*)d_in[0];
    const float* w_sam     = (const float*)d_in[1];
    const float* bn_s_sc   = (const float*)d_in[2];
    const float* bn_s_bi   = (const float*)d_in[3];
    const float* bn_s_mn   = (const float*)d_in[4];
    const float* bn_s_vr   = (const float*)d_in[5];
    const float* w_cam     = (const float*)d_in[6];
    const float* bn_c_sc   = (const float*)d_in[7];
    const float* bn_c_bi   = (const float*)d_in[8];
    const float* bn_c_mn   = (const float*)d_in[9];
    const float* bn_c_vr   = (const float*)d_in[10];
    const float* w_qk      = (const float*)d_in[11];
    const float* w_v       = (const float*)d_in[12];
    const float* gamma_sam = (const float*)d_in[13];
    const float* gamma_cam = (const float*)d_in[14];
    const float* w_out     = (const float*)d_in[15];
    (void)in_sizes; (void)n_in; (void)out_size; (void)ws_size;

    char* base = (char*)d_ws;
    size_t off = 0;
    auto carve = [&](size_t bytes) -> void* {
        off = (off + 255) & ~(size_t)255;
        void* p = base + off;
        off += bytes;
        return p;
    };

    const size_t BS = 16384;  // B * H * W
    __bf16* zpage   = (__bf16*)carve(4096);  // zero page for boundary lanes
    __bf16* xb      = (__bf16*)carve(BS * 512 * 2);
    __bf16* wp_sam  = (__bf16*)carve((size_t)9 * 512 * 512 * 2);
    __bf16* wp_cam  = (__bf16*)carve((size_t)9 * 512 * 512 * 2);
    __bf16* wp_out  = (__bf16*)carve((size_t)9 * 512 * 1024 * 2);
    __bf16* wqk_bf  = (__bf16*)carve((size_t)128 * 512 * 2);
    __bf16* wv_bf   = (__bf16*)carve((size_t)512 * 512 * 2);
    __bf16* xs_bn   = (__bf16*)carve(BS * 512 * 2);
    __bf16* xc_bn   = (__bf16*)carve(BS * 512 * 2);
    __bf16* qkbuf   = (__bf16*)carve(BS * 128 * 2);
    __bf16* vT      = (__bf16*)carve(BS * 512 * 2);
    float*  rmax    = (float*) carve(BS * 4);
    float*  rsum    = (float*) carve(BS * 4);
    __bf16* xs_out  = (__bf16*)carve(BS * 512 * 2);
    __bf16* xcT     = (__bf16*)carve(BS * 512 * 2);
    float*  mgram   = (float*) carve((size_t)4 * 512 * 512 * 4);
    __bf16* Pcam    = (__bf16*)carve((size_t)4 * 512 * 512 * 2);
    __bf16* xc_out  = (__bf16*)carve(BS * 512 * 2);

    // 0) clear zero page (ws is not re-poisoned between replays, but stay deterministic)
    k_zero_bytes<<<4, 256, 0, stream>>>((float*)zpage, 1024);

    // 1) layout conversions / weight packing
    {
        int tot = (int)(BS * 512);
        k_nchw_to_nhwc<<<(tot + 255) / 256, 256, 0, stream>>>(x, xb, 512, 4096, tot);
    }
    {
        int tot = 9 * 512 * 512;
        k_pack_conv3<<<(tot + 255) / 256, 256, 0, stream>>>(w_sam, wp_sam, 512, 512, tot);
        k_pack_conv3<<<(tot + 255) / 256, 256, 0, stream>>>(w_cam, wp_cam, 512, 512, tot);
        int tot2 = 9 * 512 * 1024;
        k_pack_conv3<<<(tot2 + 255) / 256, 256, 0, stream>>>(w_out, wp_out, 512, 1024, tot2);
        k_cvt_bf16<<<(128 * 512 + 255) / 256, 256, 0, stream>>>(w_qk, wqk_bf, 128 * 512);
        k_cvt_bf16<<<(512 * 512 + 255) / 256, 256, 0, stream>>>(w_v, wv_bf, 512 * 512);
    }

    // 2) the two 3x3 convs + BN + ReLU (bf16 NHWC out)
    k_conv3<<<dim3(4, 256), 128, 0, stream>>>(xb, 512, (const __bf16*)nullptr, 0, wp_sam,
                                              512, 0, zpage, bn_s_sc, bn_s_bi, bn_s_mn,
                                              bn_s_vr, xs_bn, (float*)nullptr);
    k_conv3<<<dim3(4, 256), 128, 0, stream>>>(xb, 512, (const __bf16*)nullptr, 0, wp_cam,
                                              512, 0, zpage, bn_c_sc, bn_c_bi, bn_c_mn,
                                              bn_c_vr, xc_bn, (float*)nullptr);

    // 3) SAM: qk and v projections, flash-softmax attention
    k_gemm1<<<dim3(1, 256), 128, 0, stream>>>(xs_bn, wqk_bf, 128, 0, qkbuf);
    k_gemm1<<<dim3(4, 256), 128, 0, stream>>>(xs_bn, wv_bf, 512, 1, vT);
    k_sam_p1<<<dim3(64, 1, 4), 128, 0, stream>>>(qkbuf, rmax, rsum);
    k_sam_p2<<<dim3(256, 1, 4), 128, 0, stream>>>(qkbuf, vT, rmax, rsum, xs_bn, gamma_sam,
                                                  xs_out);

    // 4) CAM: gram + softmax + apply
    {
        int tot = (int)(BS * 512);
        k_transpose<<<(tot + 255) / 256, 256, 0, stream>>>(xc_bn, xcT, 512, 4096, tot);
    }
    k_gram<<<dim3(4, 8, 4), 128, 0, stream>>>(xcT, mgram);
    k_softmax512<<<4 * 512, 256, 0, stream>>>(mgram, Pcam);
    k_cam_apply<<<dim3(32, 8, 4), 128, 0, stream>>>(Pcam, xc_bn, gamma_cam, xc_out);

    // 5) final 3x3 conv over virtual concat [xs_out ; xc_out] -> fp32 NCHW d_out
    k_conv3<<<dim3(4, 256), 128, 0, stream>>>(xs_out, 512, xc_out, 512, wp_out, 512, 1,
                                              zpage, (const float*)nullptr,
                                              (const float*)nullptr, (const float*)nullptr,
                                              (const float*)nullptr, (__bf16*)nullptr,
                                              (float*)d_out);
}
